// OptimizedSuperPointContrastiveLoss_7146825581131
// MI455X (gfx1250) — matrix operand order repaired and checked by reading, not compiled, not run on hardware
//
#include <hip/hip_runtime.h>
#include <hip/hip_bf16.h>
#include <math.h>

typedef __attribute__((ext_vector_type(16))) _Float16 v16h;
typedef __attribute__((ext_vector_type(8)))  float    v8f;

#define D 64
#define NCLS 20
#define TOPK 10
#define TEMP_INV 10.0f
#define NEG_INF (-__builtin_inff())
#define MAIN_WAVES 4   // 4 waves/block, 16 rows/wave -> 64 rows/block

// ---------------- vote: counts[s*20+c] += 1 ----------------
__global__ void spc_vote_kernel(const int* __restrict__ idx, const int* __restrict__ lab,
                                int* __restrict__ counts, int N) {
    int i = blockIdx.x * 256 + threadIdx.x;
    if (i < N) atomicAdd(&counts[idx[i] * NCLS + lab[i]], 1);
}

// ---------------- argmax over 20 classes ----------------
__global__ void spc_argmax_kernel(const int* __restrict__ counts, int* __restrict__ slabel, int S) {
    int s = blockIdx.x * 256 + threadIdx.x;
    if (s >= S) return;
    const int* c = counts + s * NCLS;
    int best = 0, bc = c[0];
    #pragma unroll
    for (int k = 1; k < NCLS; ++k) { int v = c[k]; if (v > bc) { bc = v; best = k; } }
    slabel[s] = best;
}

// ---------------- normalize superpoints -> f16 ----------------
__global__ __launch_bounds__(256) void spc_norm_sp_kernel(const float* __restrict__ sp,
                                                          _Float16* __restrict__ sph, int S) {
    int row  = blockIdx.x * 8 + (threadIdx.x >> 5);
    int lane = threadIdx.x & 31;
    if (row >= S) return;
    const float* p = sp + (size_t)row * D;
    float x0 = p[lane * 2], x1 = p[lane * 2 + 1];
    float s = x0 * x0 + x1 * x1;
    #pragma unroll
    for (int off = 16; off; off >>= 1) s += __shfl_xor(s, off, 32);
    float inv = 1.0f / fmaxf(sqrtf(s), 1e-12f);
    sph[(size_t)row * D + lane * 2]     = (_Float16)(x0 * inv);
    sph[(size_t)row * D + lane * 2 + 1] = (_Float16)(x1 * inv);
}

// ---------------- inverse norms for raw points ----------------
__global__ __launch_bounds__(256) void spc_inv_raw_kernel(const float* __restrict__ raw,
                                                          float* __restrict__ invn, int N) {
    int row  = blockIdx.x * 8 + (threadIdx.x >> 5);
    int lane = threadIdx.x & 31;
    if (row >= N) return;
    const float* p = raw + (size_t)row * D;
    float x0 = p[lane * 2], x1 = p[lane * 2 + 1];
    float s = x0 * x0 + x1 * x1;
    #pragma unroll
    for (int off = 16; off; off >>= 1) s += __shfl_xor(s, off, 32);
    if (lane == 0) invn[row] = 1.0f / fmaxf(sqrtf(s), 1e-12f);
}

// ---------------- main: WMMA logits + masked top-10 + loss ----------------
__global__ __launch_bounds__(128) void spc_main_kernel(
    const float* __restrict__ raw, const _Float16* __restrict__ sph,
    const float* __restrict__ invn, const int* __restrict__ idx,
    const int* __restrict__ slabel, float* __restrict__ partial, int S) {

    __shared__ float cand[MAIN_WAVES][16][16][TOPK];  // [wave][row][col-lane][k]
    __shared__ float posb[MAIN_WAVES][16];
    __shared__ float lossb[MAIN_WAVES][16];

    const int tid  = threadIdx.x;
    const int wave = tid >> 5;
    const int lane = tid & 31;
    const int hi   = lane >> 4;   // half-wave select
    const int ln   = lane & 15;
    const int m0   = (blockIdx.x * MAIN_WAVES + wave) * 16;

    // ---- A matrix: one raw row per lane, contiguous k order (k = 16*hi + s), same as B.
    // lane<16: halves 0..15 = K0..15 ; lane>=16: K16..31.
    const int rowA = m0 + ln;
    const float inv = invn[rowA];
    const float* rp = raw + (size_t)rowA * D;
    v16h a0, a1;
    #pragma unroll
    for (int j = 0; j < 8; ++j) {
        a0[j]     = (_Float16)(rp[hi * 16 + j]          * inv);
        a0[j + 8] = (_Float16)(rp[hi * 16 + 8 + j]      * inv);
        a1[j]     = (_Float16)(rp[32 + hi * 16 + j]     * inv);
        a1[j + 8] = (_Float16)(rp[32 + hi * 16 + 8 + j] * inv);
    }

    // ---- per-lane C-row state: rows m0 + hi*8 + r
    int idxr[8], labr[8];
    #pragma unroll
    for (int r = 0; r < 8; ++r) {
        int m = m0 + hi * 8 + r;
        idxr[r] = idx[m];
        labr[r] = slabel[idxr[r]];
    }
    float top[8][TOPK];
    float posv[8];
    int   hitr[8];
    #pragma unroll
    for (int r = 0; r < 8; ++r) {
        posv[r] = NEG_INF; hitr[r] = 0;
        #pragma unroll
        for (int i = 0; i < TOPK; ++i) top[r][i] = NEG_INF;
    }

    // ---- sweep all superpoint columns in 16-wide tiles
    for (int n0 = 0; n0 < S; n0 += 16) {
        const int col = n0 + ln;
        // B layout: lane<16 holds col's K run [kk .. kk+15]; lane>=16 -> +16
        const _Float16* bp = sph + (size_t)col * D + hi * 16;
        v16h b0 = *(const v16h*)(bp);
        v16h b1 = *(const v16h*)(bp + 32);
        int slab = slabel[col];

        v8f c = {0.f, 0.f, 0.f, 0.f, 0.f, 0.f, 0.f, 0.f};
        c = __builtin_amdgcn_wmma_f32_16x16x32_f16(false, a0, false, b0, (short)0, c, false, false);
        c = __builtin_amdgcn_wmma_f32_16x16x32_f16(false, a1, false, b1, (short)0, c, false, false);

        #pragma unroll
        for (int r = 0; r < 8; ++r) {
            float val = c[r] * TEMP_INV;
            bool self = (col == idxr[r]);
            if (self) { posv[r] = val; hitr[r] = 1; }
            if (self || slab == labr[r]) val = NEG_INF;
            // branchless sorted-descending insertion cascade
            #pragma unroll
            for (int i = 0; i < TOPK; ++i) {
                float mx = fmaxf(top[r][i], val);
                val      = fminf(top[r][i], val);
                top[r][i] = mx;
            }
        }
    }

    // ---- spill candidates to LDS, merge 16 per-lane lists per row
    #pragma unroll
    for (int r = 0; r < 8; ++r) {
        int row = hi * 8 + r;
        #pragma unroll
        for (int i = 0; i < TOPK; ++i) cand[wave][row][ln][i] = top[r][i];
        if (hitr[r]) posb[wave][row] = posv[r];
    }
    __syncthreads();

    if (lane < 16) {
        int row = lane;
        float t[TOPK];
        #pragma unroll
        for (int i = 0; i < TOPK; ++i) t[i] = NEG_INF;
        for (int l = 0; l < 16; ++l) {
            #pragma unroll
            for (int i = 0; i < TOPK; ++i) {
                float val = cand[wave][row][l][i];
                #pragma unroll
                for (int k = 0; k < TOPK; ++k) {
                    float mx = fmaxf(t[k], val);
                    val      = fminf(t[k], val);
                    t[k] = mx;
                }
            }
        }
        float neg = 0.f;
        #pragma unroll
        for (int i = 0; i < TOPK; ++i) neg += expf(t[i]);
        float ep = expf(posb[wave][row]);
        lossb[wave][row] = -logf(ep / (ep + neg + 1e-8f));
    }
    __syncthreads();

    if (tid == 0) {
        float s = 0.f;
        for (int w = 0; w < MAIN_WAVES; ++w)
            for (int i = 0; i < 16; ++i) s += lossb[w][i];
        partial[blockIdx.x] = s;
    }
}

// ---------------- deterministic final reduction ----------------
__global__ __launch_bounds__(256) void spc_final_kernel(const float* __restrict__ partial,
                                                        float* __restrict__ out,
                                                        int nb, float scale) {
    __shared__ float s[256];
    float acc = 0.f;
    for (int i = threadIdx.x; i < nb; i += 256) acc += partial[i];
    s[threadIdx.x] = acc;
    __syncthreads();
    for (int off = 128; off; off >>= 1) {
        if (threadIdx.x < off) s[threadIdx.x] += s[threadIdx.x + off];
        __syncthreads();
    }
    if (threadIdx.x == 0) out[0] = s[0] * scale;
}

static inline size_t align256(size_t x) { return (x + 255) & ~(size_t)255; }

extern "C" void kernel_launch(void* const* d_in, const int* in_sizes, int n_in,
                              void* d_out, int out_size, void* d_ws, size_t ws_size,
                              hipStream_t stream) {
    const float* sp  = (const float*)d_in[0];   // [S, 64]
    const float* raw = (const float*)d_in[1];   // [N, 64]
    const int*   idx = (const int*)d_in[2];     // [N]
    const int*   lab = (const int*)d_in[3];     // [N]
    const int S = in_sizes[0] / D;
    const int N = in_sizes[1] / D;

    char* ws = (char*)d_ws;
    _Float16* sph    = (_Float16*)ws;  ws += align256((size_t)S * D * sizeof(_Float16));
    float*    invn   = (float*)ws;     ws += align256((size_t)N * sizeof(float));
    int*      counts = (int*)ws;       ws += align256((size_t)S * NCLS * sizeof(int));
    int*      slabel = (int*)ws;       ws += align256((size_t)S * sizeof(int));
    float*    partial= (float*)ws;

    const int nblocks = N / (MAIN_WAVES * 16);

    hipMemsetAsync(counts, 0, (size_t)S * NCLS * sizeof(int), stream);
    spc_vote_kernel<<<(N + 255) / 256, 256, 0, stream>>>(idx, lab, counts, N);
    spc_argmax_kernel<<<(S + 255) / 256, 256, 0, stream>>>(counts, slabel, S);
    spc_norm_sp_kernel<<<(S + 7) / 8, 256, 0, stream>>>(sp, sph, S);
    spc_inv_raw_kernel<<<(N + 7) / 8, 256, 0, stream>>>(raw, invn, N);
    spc_main_kernel<<<nblocks, 128, 0, stream>>>(raw, sph, invn, idx, slabel, partial, S);
    // total = (sum of per-row losses)/N * LOSS_WEIGHT * TOTAL_WEIGHT
    spc_final_kernel<<<1, 256, 0, stream>>>(partial, (float*)d_out, nblocks, 0.1f / (float)N);
}
